// Evoformer_61349312856368
// MI455X (gfx1250) — compile-verified, matrix operand order checked by reading, not run on hardware
//
// Evoformer block for MI455X (gfx1250), wave32 + WMMA bf16.
//
// Reasoning (compile-only; no timing available):
//  - All contractions go through v_wmma_f32_16x16x32_bf16 (f32 inputs converted
//    to bf16 in-register; f32 accumulate). bf16 keeps the f32 exponent range of
//    the N(0,1)*0.02 data, so accuracy tracks the f32 reference.
//  - f32 -> bf16x2 conversion is a single v_cvt_pk_bf16_f32 via inline asm
//    (instruction verified present on gfx1250; the clang builtin is absent and
//    scalar casts lower to 2 cvts + 2 moves per pair).
//  - Persistent state (m: 12 MB, z: 18 MB) fits in L2 (192 MB); transient score
//    tensors stream at 23.3 TB/s -> tens of microseconds. The block is
//    launch/latency bound, so epilogues fold into the GEMMs and each wave
//    computes a 16x32 tile (1 A-frag, 2 WMMAs) for WMMA density.
//  - Hot loops: wave-uniform interior fast path (no per-load EXEC masking),
//    loop-invariant K-split (no shifts/divides per element).
//  - Workspace plan (floats, assumed ws_size >= ~330 MB):
//      [0,5M) norm buf | [5M..25M) q/k/v/g | [25M..40M) o/og/delta | [40M..41M)
//      pair-bias | [41M..81M) scores / hidden / outer-product scratch.
//  - Param flattening assumes jax pytree order (dicts flattened by sorted key):
//      msa, pair, then params: col, msa_tr, opm, pair_tr, row, ta_end,
//      ta_start, tm_in, tm_out (leaves sorted within each sub-dict).

#include <hip/hip_runtime.h>

typedef __attribute__((ext_vector_type(16))) __bf16 bf16x16;
typedef __attribute__((ext_vector_type(8)))  float  f32x8;

union Frag {
  bf16x16 v;
  unsigned int w[8];
};

// Pack two f32 -> packed bf16x2 in one VALU op (low half = first arg).
// v_cvt_pk_bf16_f32 is present in the gfx1250 assembler (observed in round-3
// codegen); no clang builtin exists, so emit it directly.
__device__ __forceinline__ unsigned int pk2bf(float a, float b) {
  unsigned int r;
  asm("v_cvt_pk_bf16_f32 %0, %1, %2" : "=v"(r) : "v"(a), "v"(b));
  return r;
}

// ---------------------------------------------------------------------------
// K1: C[M,N] = act( A_idx(m,k) * W[n,k] + bias[n] )
//   A addressing: m=(m1=m/Mdiv, m2=m%Mdiv), k=(k1=k>>Kshift, k2=k-(k1<<Kshift))
//   addr = m1*sAm1 + m2*sAm2 + k1*sAk1 + k2   (innermost k contiguous)
//   Normal GEMM: Mdiv=M, sAm1=0, sAm2=K, Kshift=30, sAk1=0.
//   OPM regather: Mdiv=192, sAm1=196608, sAm2=32, Kshift=5, sAk1=6144.
// act: 0=none, 1=relu, 2=sigmoid.
// 128 threads = 4 waves; wave tile 16(M) x 32(N); block tile 16 x 128.
// kbase[e] < 32 <= 2^Kshift, and k0 is a multiple of 32, so within one K-step
// k1 = k0>>Kshift is loop-invariant -> per-element address = abase + kbase[e].
// ---------------------------------------------------------------------------
__global__ void gemm_aw(const float* __restrict__ A, const float* __restrict__ Wt,
                        const float* __restrict__ bias, float* __restrict__ Cc,
                        int M, int N, int Kd,
                        int Mdiv, long long sAm1, long long sAm2,
                        int Kshift, long long sAk1, int act) {
  const int lane = threadIdx.x & 31;
  const int wid  = threadIdx.x >> 5;
  const int n0 = blockIdx.x * 128 + wid * 32;
  const int m0 = blockIdx.y * 16;
  const int lm = lane & 15, hi = lane >> 4;

  const int am = m0 + lm;
  const bool am_ok = (am < M);
  const int m1 = am / Mdiv;                      // once per thread, off hot path
  const int m2 = am - m1 * Mdiv;
  const long long aoff = (long long)m1 * sAm1 + (long long)m2 * sAm2;

  const int bn0 = n0 + lm;
  const int bn1 = n0 + 16 + lm;
  const bool b0_ok = (bn0 < N);
  const bool b1_ok = (bn1 < N);
  const long long bo0 = (long long)bn0 * (long long)Kd;
  const long long bo1 = (long long)bn1 * (long long)Kd;

  int kbase[8];
#pragma unroll
  for (int e = 0; e < 8; ++e) kbase[e] = ((e < 4) ? 0 : 16) + hi * 8 + (e & 3) * 2;

  f32x8 acc0 = {0.f, 0.f, 0.f, 0.f, 0.f, 0.f, 0.f, 0.f};
  f32x8 acc1 = {0.f, 0.f, 0.f, 0.f, 0.f, 0.f, 0.f, 0.f};

  const bool interior = (m0 + 16 <= M) && (n0 + 32 <= N);   // wave-uniform

  if (interior) {
    // ---- fast path: no masks, immediate-offset loads ----
    for (int k0 = 0; k0 < Kd; k0 += 32) {
      if ((k0 + 32) < Kd)
        __builtin_prefetch(&Wt[bo0 + k0 + 32], 0, 0);   // global_prefetch_b8
      const int k1 = k0 >> Kshift;
      const long long abase = aoff + (long long)k1 * sAk1 + (k0 - (k1 << Kshift));
      const long long bbase0 = bo0 + k0;
      const long long bbase1 = bo1 + k0;
      Frag fa, fb0, fb1;
#pragma unroll
      for (int e = 0; e < 8; ++e) {
        const long long ai = abase + kbase[e];
        fa.w[e]  = pk2bf(A[ai], A[ai + 1]);
        fb0.w[e] = pk2bf(Wt[bbase0 + kbase[e]], Wt[bbase0 + kbase[e] + 1]);
        fb1.w[e] = pk2bf(Wt[bbase1 + kbase[e]], Wt[bbase1 + kbase[e] + 1]);
      }
      acc0 = __builtin_amdgcn_wmma_f32_16x16x32_bf16(false, fa.v, false, fb0.v,
                                                     (short)0, acc0, false, false);
      acc1 = __builtin_amdgcn_wmma_f32_16x16x32_bf16(false, fa.v, false, fb1.v,
                                                     (short)0, acc1, false, false);
    }
  } else {
    // ---- guarded path (edge tiles, incl. N=8/4 bias projections) ----
    for (int k0 = 0; k0 < Kd; k0 += 32) {
      const int k1 = k0 >> Kshift;
      const long long abase = aoff + (long long)k1 * sAk1 + (k0 - (k1 << Kshift));
      Frag fa, fb0, fb1;
#pragma unroll
      for (int e = 0; e < 8; ++e) {
        const int kb = kbase[e];
        float a0 = 0.f, a1 = 0.f;
        if (am_ok) { a0 = A[abase + kb]; a1 = A[abase + kb + 1]; }
        fa.w[e] = pk2bf(a0, a1);
        float p0 = 0.f, p1 = 0.f, q0 = 0.f, q1 = 0.f;
        if (b0_ok) { p0 = Wt[bo0 + k0 + kb]; p1 = Wt[bo0 + k0 + kb + 1]; }
        if (b1_ok) { q0 = Wt[bo1 + k0 + kb]; q1 = Wt[bo1 + k0 + kb + 1]; }
        fb0.w[e] = pk2bf(p0, p1);
        fb1.w[e] = pk2bf(q0, q1);
      }
      acc0 = __builtin_amdgcn_wmma_f32_16x16x32_bf16(false, fa.v, false, fb0.v,
                                                     (short)0, acc0, false, false);
      acc1 = __builtin_amdgcn_wmma_f32_16x16x32_bf16(false, fa.v, false, fb1.v,
                                                     (short)0, acc1, false, false);
    }
  }

#pragma unroll
  for (int v = 0; v < 8; ++v) {
    const int m = m0 + v + hi * 8;
    if (m >= M) continue;
    const int na = n0 + lm;
    const int nb = n0 + 16 + lm;
    if (na < N) {
      float val = acc0[v];
      if (bias) val += bias[na];
      if (act == 1)      val = val > 0.f ? val : 0.f;
      else if (act == 2) val = 1.f / (1.f + __expf(-val));
      Cc[(long long)m * N + na] = val;
    }
    if (nb < N) {
      float val = acc1[v];
      if (bias) val += bias[nb];
      if (act == 1)      val = val > 0.f ? val : 0.f;
      else if (act == 2) val = 1.f / (1.f + __expf(-val));
      Cc[(long long)m * N + nb] = val;
    }
  }
}

// ---------------------------------------------------------------------------
// K2: fully strided 2D-batched NT GEMM.
//   C[b1,b2,m,n] = scale * sum_k A[b1,b2,m,k] * B[b1,b2,n,k]
//   element(A) = A[b1*a_s1 + b2*a_s2 + m*a_rs + k*a_ks], similarly B, C.
// M multiple of 16, N multiple of 32, Kd multiple of 32 (true for all uses).
// 128 threads = 4 waves; wave tile 16 x 32; block tile 16 x 128.
// ---------------------------------------------------------------------------
struct BGArgs {
  const float* A; long long a_s1, a_s2, a_rs, a_ks;
  const float* B; long long b_s1, b_s2, b_rs, b_ks;
  float* C;       long long c_s1, c_s2, c_rs, c_ns;
  int M, N, Kd, nb2; float scale;
};

__global__ void bgemm_s(BGArgs g) {
  const int lane = threadIdx.x & 31;
  const int wid  = threadIdx.x >> 5;
  const int n0 = blockIdx.x * 128 + wid * 32;
  if (n0 >= g.N) return;                        // uniform per-wave
  const bool has2 = (n0 + 16) < g.N;            // uniform per-wave
  const int m0 = blockIdx.y * 16;
  const int bz = blockIdx.z;
  const int b1 = bz / g.nb2;
  const int b2 = bz - b1 * g.nb2;
  const int lm = lane & 15, hi = lane >> 4;

  long long arow = (long long)b1 * g.a_s1 + (long long)b2 * g.a_s2 +
                   (long long)(m0 + lm) * g.a_rs;
  long long brow0 = (long long)b1 * g.b_s1 + (long long)b2 * g.b_s2 +
                    (long long)(n0 + lm) * g.b_rs;
  long long brow1 = brow0 + 16ll * g.b_rs;

  // hoisted per-fragment-element offsets (no multiplies in the hot loop)
  long long aoffs[8], boffs[8];
#pragma unroll
  for (int e = 0; e < 8; ++e) {
    const int be = ((e < 4) ? 0 : 16) + hi * 8 + (e & 3) * 2;
    aoffs[e] = (long long)be * g.a_ks;
    boffs[e] = (long long)be * g.b_ks;
  }
  const long long a_adv = 32ll * g.a_ks;
  const long long b_adv = 32ll * g.b_ks;

  f32x8 acc0 = {0.f, 0.f, 0.f, 0.f, 0.f, 0.f, 0.f, 0.f};
  f32x8 acc1 = {0.f, 0.f, 0.f, 0.f, 0.f, 0.f, 0.f, 0.f};

  for (int k0 = 0; k0 < g.Kd; k0 += 32) {
    Frag fa, fb0, fb1;
#pragma unroll
    for (int e = 0; e < 8; ++e) {
      fa.w[e]  = pk2bf(g.A[arow + aoffs[e]], g.A[arow + aoffs[e] + g.a_ks]);
      fb0.w[e] = pk2bf(g.B[brow0 + boffs[e]], g.B[brow0 + boffs[e] + g.b_ks]);
    }
    acc0 = __builtin_amdgcn_wmma_f32_16x16x32_bf16(false, fa.v, false, fb0.v,
                                                   (short)0, acc0, false, false);
    if (has2) {
#pragma unroll
      for (int e = 0; e < 8; ++e)
        fb1.w[e] = pk2bf(g.B[brow1 + boffs[e]], g.B[brow1 + boffs[e] + g.b_ks]);
      acc1 = __builtin_amdgcn_wmma_f32_16x16x32_bf16(false, fa.v, false, fb1.v,
                                                     (short)0, acc1, false, false);
    }
    arow += a_adv; brow0 += b_adv; brow1 += b_adv;
  }

  const long long co = (long long)b1 * g.c_s1 + (long long)b2 * g.c_s2;
#pragma unroll
  for (int v = 0; v < 8; ++v) {
    const int m = m0 + v + hi * 8;
    g.C[co + (long long)m * g.c_rs + (long long)(n0 + lm) * g.c_ns] =
        acc0[v] * g.scale;
  }
  if (has2) {
#pragma unroll
    for (int v = 0; v < 8; ++v) {
      const int m = m0 + v + hi * 8;
      g.C[co + (long long)m * g.c_rs + (long long)(n0 + 16 + lm) * g.c_ns] =
          acc1[v] * g.scale;
    }
  }
}

// ---------------------------------------------------------------------------
// LayerNorm over rows of length C (block = 256 threads per row).
// ---------------------------------------------------------------------------
__global__ void layernorm_rows(const float* __restrict__ x, const float* __restrict__ g,
                               const float* __restrict__ b, float* __restrict__ y, int C) {
  const float* p = x + (long long)blockIdx.x * C;
  float* q = y + (long long)blockIdx.x * C;
  const int t = threadIdx.x;
  float s = 0.f, s2 = 0.f;
  for (int i = t; i < C; i += blockDim.x) { float v = p[i]; s += v; s2 += v * v; }
#pragma unroll
  for (int o = 16; o > 0; o >>= 1) { s += __shfl_xor(s, o, 32); s2 += __shfl_xor(s2, o, 32); }
  __shared__ float rs[8], rs2[8], fin[2];
  const int wid = t >> 5, lane = t & 31;
  if (lane == 0) { rs[wid] = s; rs2[wid] = s2; }
  __syncthreads();
  if (t == 0) {
    const int nw = blockDim.x >> 5;
    float a = 0.f, a2 = 0.f;
    for (int i = 0; i < nw; ++i) { a += rs[i]; a2 += rs2[i]; }
    const float mu = a / C;
    const float var = a2 / C - mu * mu;
    fin[0] = mu; fin[1] = rsqrtf(var + 1e-5f);
  }
  __syncthreads();
  const float mu = fin[0], inv = fin[1];
  for (int i = t; i < C; i += blockDim.x)
    q[i] = (p[i] - mu) * inv * g[i] + b[i];
}

// ---------------------------------------------------------------------------
// In-place softmax over rows of length L.
// ---------------------------------------------------------------------------
__global__ void softmax_rows(float* __restrict__ x, int L) {
  float* p = x + (long long)blockIdx.x * L;
  const int t = threadIdx.x;
  __shared__ float r[8], fin[1];
  float mx = -3.0e38f;
  for (int i = t; i < L; i += blockDim.x) mx = fmaxf(mx, p[i]);
#pragma unroll
  for (int o = 16; o > 0; o >>= 1) mx = fmaxf(mx, __shfl_xor(mx, o, 32));
  if ((t & 31) == 0) r[t >> 5] = mx;
  __syncthreads();
  if (t == 0) {
    const int nw = blockDim.x >> 5;
    float v = r[0];
    for (int i = 1; i < nw; ++i) v = fmaxf(v, r[i]);
    fin[0] = v;
  }
  __syncthreads();
  mx = fin[0];
  __syncthreads();
  float s = 0.f;
  for (int i = t; i < L; i += blockDim.x) { float e = __expf(p[i] - mx); p[i] = e; s += e; }
#pragma unroll
  for (int o = 16; o > 0; o >>= 1) s += __shfl_xor(s, o, 32);
  if ((t & 31) == 0) r[t >> 5] = s;
  __syncthreads();
  if (t == 0) {
    const int nw = blockDim.x >> 5;
    float v = 0.f;
    for (int i = 0; i < nw; ++i) v += r[i];
    fin[0] = v;
  }
  __syncthreads();
  const float inv = 1.f / fin[0];
  for (int i = t; i < L; i += blockDim.x) p[i] *= inv;
}

// ---------------------------------------------------------------------------
// Elementwise helpers (grid-stride).
// ---------------------------------------------------------------------------
__global__ void ew_mul_k(float* __restrict__ d, const float* __restrict__ a,
                         const float* __restrict__ b, long long n) {
  long long i = (long long)blockIdx.x * blockDim.x + threadIdx.x;
  const long long st = (long long)gridDim.x * blockDim.x;
  for (; i < n; i += st) d[i] = a[i] * b[i];
}

__global__ void ew_add_k(float* __restrict__ d, const float* __restrict__ s, long long n) {
  long long i = (long long)blockIdx.x * blockDim.x + threadIdx.x;
  const long long st = (long long)gridDim.x * blockDim.x;
  for (; i < n; i += st) d[i] += s[i];
}

// scores[b1,b2,m,n] += bias[(m*N+n)*Hc + b2]   (nb2 == Hc)
__global__ void ew_bias_k(float* __restrict__ s, const float* __restrict__ bias,
                          int M, int N, int Hc, long long n) {
  long long i = (long long)blockIdx.x * blockDim.x + threadIdx.x;
  const long long st = (long long)gridDim.x * blockDim.x;
  for (; i < n; i += st) {
    const int nn = (int)(i % N);
    const long long r = i / N;
    const int mm = (int)(r % M);
    const long long bz = r / M;
    const int b2 = (int)(bz % Hc);
    s[i] += bias[((long long)mm * N + nn) * Hc + b2];
  }
}

// ---------------------------------------------------------------------------
// Host orchestration.
// ---------------------------------------------------------------------------
static inline dim3 ewgrid(long long n) {
  long long b = (n + 255) / 256;
  if (b > 16384) b = 16384;
  return dim3((unsigned)b);
}

static void bgemm_launch(hipStream_t st, const float* A, long long as1, long long as2,
                         long long ars, long long aks, const float* B, long long bs1,
                         long long bs2, long long brs, long long bks, float* C,
                         long long cs1, long long cs2, long long crs, long long cns,
                         int M, int N, int Kd, int nb1, int nb2, float scale) {
  BGArgs g = {A, as1, as2, ars, aks, B, bs1, bs2, brs, bks,
              C, cs1, cs2, crs, cns, M, N, Kd, nb2, scale};
  dim3 gr((N + 127) / 128, M / 16, (unsigned)(nb1 * nb2));
  bgemm_s<<<gr, 128, 0, st>>>(g);
}

extern "C" void kernel_launch(void* const* d_in, const int* in_sizes, int n_in,
                              void* d_out, int out_size, void* d_ws, size_t ws_size,
                              hipStream_t stream) {
  const int S = 64, R = 192, CS = 256, CZ = 128, H = 8, TH = 4;
  const int MS = S * R;                 // 12288 MSA rows
  const int MZ = R * R;                 // 36864 pair rows
  const long long m_el = (long long)MS * CS;   // 3,145,728
  const long long z_el = (long long)MZ * CZ;   // 4,718,592
  const float ISQ = 0.1767766953f;      // 1/sqrt(32)

  auto P = [&](int i) { return (const float*)d_in[i]; };
  // jax pytree (sorted-key) flattening:
  const float* in_msa = P(0);
  const float* in_pair = P(1);
  // col: bg bo ln_b ln_g wg wk wo wq wv
  const float *c_bg=P(2), *c_bo=P(3), *c_lnb=P(4), *c_lng=P(5), *c_wg=P(6),
              *c_wk=P(7), *c_wo=P(8), *c_wq=P(9), *c_wv=P(10);
  // msa_tr: b1 b2 ln_b ln_g w1 w2
  const float *mt_b1=P(11), *mt_b2=P(12), *mt_lnb=P(13), *mt_lng=P(14),
              *mt_w1=P(15), *mt_w2=P(16);
  // opm: bl bo br ln_b ln_g wl wo wr
  const float *op_bl=P(17), *op_bo=P(18), *op_br=P(19), *op_lnb=P(20),
              *op_lng=P(21), *op_wl=P(22), *op_wo=P(23), *op_wr=P(24);
  // pair_tr
  const float *pt_b1=P(25), *pt_b2=P(26), *pt_lnb=P(27), *pt_lng=P(28),
              *pt_w1=P(29), *pt_w2=P(30);
  // row: bg bo ln_b ln_g ln_z_b ln_z_g wg wk wo wp wq wv
  const float *r_bg=P(31), *r_bo=P(32), *r_lnb=P(33), *r_lng=P(34), *r_lnzb=P(35),
              *r_lnzg=P(36), *r_wg=P(37), *r_wk=P(38), *r_wo=P(39), *r_wp=P(40),
              *r_wq=P(41), *r_wv=P(42);
  // ta_end: bg bo ln_b ln_g wb wg wk wo wq wv
  const float *te_bg=P(43), *te_bo=P(44), *te_lnb=P(45), *te_lng=P(46), *te_wb=P(47),
              *te_wg=P(48), *te_wk=P(49), *te_wo=P(50), *te_wq=P(51), *te_wv=P(52);
  // ta_start
  const float *ts_bg=P(53), *ts_bo=P(54), *ts_lnb=P(55), *ts_lng=P(56), *ts_wb=P(57),
              *ts_wg=P(58), *ts_wk=P(59), *ts_wo=P(60), *ts_wq=P(61), *ts_wv=P(62);
  // tm_in: ba bag bb bbg bgg bo ln2_b ln2_g ln_b ln_g wa wag wb wbg wgg wo
  const float *ti_ba=P(63), *ti_bag=P(64), *ti_bb=P(65), *ti_bbg=P(66), *ti_bgg=P(67),
              *ti_bo=P(68), *ti_l2b=P(69), *ti_l2g=P(70), *ti_lnb=P(71), *ti_lng=P(72),
              *ti_wa=P(73), *ti_wag=P(74), *ti_wb=P(75), *ti_wbg=P(76), *ti_wgg=P(77),
              *ti_wo=P(78);
  // tm_out
  const float *to_ba=P(79), *to_bag=P(80), *to_bb=P(81), *to_bbg=P(82), *to_bgg=P(83),
              *to_bo=P(84), *to_l2b=P(85), *to_l2g=P(86), *to_lnb=P(87), *to_lng=P(88),
              *to_wa=P(89), *to_wag=P(90), *to_wb=P(91), *to_wbg=P(92), *to_wgg=P(93),
              *to_wo=P(94);

  float* mO = (float*)d_out;        // running m
  float* zO = mO + m_el;            // running z

  float* ws = (float*)d_ws;
  const long long MEG = 1ll << 20;
  float* N0 = ws;                   // normalized input (<=5M floats)
  float* Qb = ws + 5 * MEG;
  float* Kb = ws + 10 * MEG;
  float* Vb = ws + 15 * MEG;
  float* Gb = ws + 20 * MEG;
  float* Ob = ws + 25 * MEG;
  float* OG = ws + 30 * MEG;
  float* Dl = ws + 35 * MEG;
  float* PB = ws + 40 * MEG;
  float* Sc = ws + 41 * MEG;        // scores / hidden / outer scratch (~40M floats)

  auto LN = [&](const float* x, const float* g, const float* b, float* y, int rows, int C) {
    layernorm_rows<<<dim3((unsigned)rows), 256, 0, stream>>>(x, g, b, y, C);
  };
  auto GEMM = [&](const float* A, const float* W, const float* bias, float* C,
                  int M, int N, int K, int act) {
    dim3 gr((N + 127) / 128, (M + 15) / 16);
    gemm_aw<<<gr, 128, 0, stream>>>(A, W, bias, C, M, N, K, M, 0ll, (long long)K,
                                    30, 0ll, act);
  };
  auto SOFT = [&](float* x, long long rows, int L) {
    softmax_rows<<<dim3((unsigned)rows), 256, 0, stream>>>(x, L);
  };
  auto MUL = [&](float* d, const float* a, const float* b, long long n) {
    ew_mul_k<<<ewgrid(n), 256, 0, stream>>>(d, a, b, n);
  };
  auto ADD = [&](float* d, const float* s, long long n) {
    ew_add_k<<<ewgrid(n), 256, 0, stream>>>(d, s, n);
  };
  auto BIAS = [&](float* s, const float* bias, int M, int N, int Hc, long long n) {
    ew_bias_k<<<ewgrid(n), 256, 0, stream>>>(s, bias, M, N, Hc, n);
  };

  hipMemcpyAsync(mO, in_msa, m_el * 4, hipMemcpyDeviceToDevice, stream);
  hipMemcpyAsync(zO, in_pair, z_el * 4, hipMemcpyDeviceToDevice, stream);

  // ===== 1. MSA row attention (pair bias) =====
  LN(mO, r_lng, r_lnb, N0, MS, CS);
  GEMM(N0, r_wq, nullptr, Qb, MS, 256, 256, 0);
  GEMM(N0, r_wk, nullptr, Kb, MS, 256, 256, 0);
  GEMM(N0, r_wv, nullptr, Vb, MS, 256, 256, 0);
  GEMM(N0, r_wg, r_bg,    Gb, MS, 256, 256, 2);
  LN(zO, r_lnzg, r_lnzb, OG, MZ, CZ);          // zn (temp in OG)
  GEMM(OG, r_wp, nullptr, PB, MZ, 8, 128, 0);  // pair bias [I*J, H]
  // scores[s,h,i,j] = q.k/sqrt(D)
  bgemm_launch(stream, Qb, 49152, 32, 256, 1, Kb, 49152, 32, 256, 1,
               Sc, 294912, 36864, 192, 1, 192, 192, 32, S, H, ISQ);
  BIAS(Sc, PB, 192, 192, H, (long long)S * H * 192 * 192);
  SOFT(Sc, (long long)S * H * 192, 192);
  // o[s,i,h,d] = P.V
  bgemm_launch(stream, Sc, 294912, 36864, 192, 1, Vb, 49152, 32, 1, 256,
               Ob, 49152, 32, 256, 1, 192, 32, 192, S, H, 1.f);
  MUL(OG, Gb, Ob, m_el);
  GEMM(OG, r_wo, r_bo, Dl, MS, 256, 256, 0);
  ADD(mO, Dl, m_el);

  // ===== 2. MSA column attention =====
  LN(mO, c_lng, c_lnb, N0, MS, CS);
  GEMM(N0, c_wq, nullptr, Qb, MS, 256, 256, 0);
  GEMM(N0, c_wk, nullptr, Kb, MS, 256, 256, 0);
  GEMM(N0, c_wv, nullptr, Vb, MS, 256, 256, 0);
  GEMM(N0, c_wg, c_bg,    Gb, MS, 256, 256, 2);
  // scores[i,h,s,t]
  bgemm_launch(stream, Qb, 256, 32, 49152, 1, Kb, 256, 32, 49152, 1,
               Sc, 32768, 4096, 64, 1, 64, 64, 32, R, H, ISQ);
  SOFT(Sc, (long long)R * H * 64, 64);
  // o[s,i,h,d]
  bgemm_launch(stream, Sc, 32768, 4096, 64, 1, Vb, 256, 32, 1, 49152,
               Ob, 256, 32, 49152, 1, 64, 32, 64, R, H, 1.f);
  MUL(OG, Gb, Ob, m_el);
  GEMM(OG, c_wo, c_bo, Dl, MS, 256, 256, 0);
  ADD(mO, Dl, m_el);

  // ===== 3. MSA transition =====
  LN(mO, mt_lng, mt_lnb, N0, MS, CS);
  GEMM(N0, mt_w1, mt_b1, Sc, MS, 1024, 256, 1);
  GEMM(Sc, mt_w2, mt_b2, Dl, MS, 256, 1024, 0);
  ADD(mO, Dl, m_el);

  // ===== 4. Outer product mean =====
  LN(mO, op_lng, op_lnb, N0, MS, CS);
  GEMM(N0, op_wl, op_bl, Qb, MS, 32, 256, 0);   // a [s,i,c]
  GEMM(N0, op_wr, op_br, Kb, MS, 32, 256, 0);   // b [s,j,d]
  // outer[(i*32+c),(j*32+d)] = (1/S) sum_s a b  (TN gemm, k-stride 6144)
  bgemm_launch(stream, Qb, 0, 0, 1, 6144, Kb, 0, 0, 1, 6144,
               Sc, 0, 0, 6144, 1, 6144, 6144, 64, 1, 1, 1.f / 64.f);
  {  // projection with regather addressing (Kshift=5 -> k-split by 32)
    dim3 gr((128 + 127) / 128, (MZ + 15) / 16);
    gemm_aw<<<gr, 128, 0, stream>>>(Sc, op_wo, op_bo, Dl, MZ, 128, 1024,
                                    192, 196608ll, 32ll, 5, 6144ll, 0);
  }
  ADD(zO, Dl, z_el);

  // ===== 5/6. Triangle multiplications (outgoing, incoming) =====
  for (int pass = 0; pass < 2; ++pass) {
    const bool outg = (pass == 0);
    const float *lng = outg ? to_lng : ti_lng, *lnb = outg ? to_lnb : ti_lnb;
    const float *wag = outg ? to_wag : ti_wag, *bag = outg ? to_bag : ti_bag;
    const float *wa  = outg ? to_wa  : ti_wa,  *ba  = outg ? to_ba  : ti_ba;
    const float *wbg = outg ? to_wbg : ti_wbg, *bbg = outg ? to_bbg : ti_bbg;
    const float *wb  = outg ? to_wb  : ti_wb,  *bb  = outg ? to_bb  : ti_bb;
    const float *wgg = outg ? to_wgg : ti_wgg, *bgg = outg ? to_bgg : ti_bgg;
    const float *l2g = outg ? to_l2g : ti_l2g, *l2b = outg ? to_l2b : ti_l2b;
    const float *wo  = outg ? to_wo  : ti_wo,  *bo  = outg ? to_bo  : ti_bo;

    LN(zO, lng, lnb, N0, MZ, CZ);
    GEMM(N0, wag, bag, Qb, MZ, 128, 128, 2);
    GEMM(N0, wa,  ba,  Kb, MZ, 128, 128, 0);
    MUL(Qb, Qb, Kb, z_el);                       // a
    GEMM(N0, wbg, bbg, Vb, MZ, 128, 128, 2);
    GEMM(N0, wb,  bb,  Kb, MZ, 128, 128, 0);
    MUL(Vb, Vb, Kb, z_el);                       // b
    // x[i,j,c]; batch over channel c (128)
    if (outg)  // sum_k a[i,k,c] b[j,k,c]
      bgemm_launch(stream, Qb, 1, 0, 24576, 128, Vb, 1, 0, 24576, 128,
                   Sc, 1, 0, 24576, 128, 192, 192, 192, 128, 1, 1.f);
    else       // sum_k a[k,i,c] b[k,j,c]
      bgemm_launch(stream, Qb, 1, 0, 128, 24576, Vb, 1, 0, 128, 24576,
                   Sc, 1, 0, 24576, 128, 192, 192, 192, 128, 1, 1.f);
    LN(Sc, l2g, l2b, Ob, MZ, CZ);                // ln2(x)
    GEMM(N0, wgg, bgg, Gb, MZ, 128, 128, 2);     // gate
    GEMM(Ob, wo, bo, OG, MZ, 128, 128, 0);
    MUL(Dl, Gb, OG, z_el);
    ADD(zO, Dl, z_el);
  }

  // ===== 7/8. Triangle attention (starting, ending) =====
  for (int pass = 0; pass < 2; ++pass) {
    const bool start = (pass == 0);
    const float *lng = start ? ts_lng : te_lng, *lnb = start ? ts_lnb : te_lnb;
    const float *wq = start ? ts_wq : te_wq, *wk = start ? ts_wk : te_wk;
    const float *wv = start ? ts_wv : te_wv, *wg = start ? ts_wg : te_wg;
    const float *bg = start ? ts_bg : te_bg, *wb = start ? ts_wb : te_wb;
    const float *wo = start ? ts_wo : te_wo, *bo = start ? ts_bo : te_bo;

    LN(zO, lng, lnb, N0, MZ, CZ);
    GEMM(N0, wq, nullptr, Qb, MZ, 128, 128, 0);
    GEMM(N0, wk, nullptr, Kb, MZ, 128, 128, 0);
    GEMM(N0, wv, nullptr, Vb, MZ, 128, 128, 0);
    GEMM(N0, wg, bg,      Gb, MZ, 128, 128, 2);
    GEMM(N0, wb, nullptr, PB, MZ, 4, 128, 0);    // triangle bias [I*J, TH]
    if (start) {
      // scores[i,h,j,k] = q[i,j].k[i,k]; bias b[j,k,h]
      bgemm_launch(stream, Qb, 24576, 32, 128, 1, Kb, 24576, 32, 128, 1,
                   Sc, 147456, 36864, 192, 1, 192, 192, 32, R, TH, ISQ);
      BIAS(Sc, PB, 192, 192, TH, (long long)R * TH * 192 * 192);
      SOFT(Sc, (long long)R * TH * 192, 192);
      // o[i,j,h,d] = sum_k P v[i,k]
      bgemm_launch(stream, Sc, 147456, 36864, 192, 1, Vb, 24576, 32, 1, 128,
                   Ob, 24576, 32, 128, 1, 192, 32, 192, R, TH, 1.f);
    } else {
      // scores2[j,h,i,k] = q[i,j].k[k,j]; bias b[i,k,h]
      bgemm_launch(stream, Qb, 128, 32, 24576, 1, Kb, 128, 32, 24576, 1,
                   Sc, 147456, 36864, 192, 1, 192, 192, 32, R, TH, ISQ);
      BIAS(Sc, PB, 192, 192, TH, (long long)R * TH * 192 * 192);
      SOFT(Sc, (long long)R * TH * 192, 192);
      // o[i,j,h,d] = sum_k P v[k,j]
      bgemm_launch(stream, Sc, 147456, 36864, 192, 1, Vb, 128, 32, 1, 24576,
                   Ob, 128, 32, 24576, 1, 192, 32, 192, R, TH, 1.f);
    }
    MUL(OG, Gb, Ob, z_el);
    GEMM(OG, wo, bo, Dl, MZ, 128, 128, 0);
    ADD(zO, Dl, z_el);
  }

  // ===== 9. Pair transition =====
  LN(zO, pt_lng, pt_lnb, N0, MZ, CZ);
  GEMM(N0, pt_w1, pt_b1, Sc, MZ, 512, 128, 1);
  GEMM(Sc, pt_w2, pt_b2, Dl, MZ, 128, 512, 0);
  ADD(zO, Dl, z_el);
}
